// PolyGIN_22290880266887
// MI455X (gfx1250) — compile-verified, matrix operand order checked by reading, not run on hardware
//
#include <hip/hip_runtime.h>

typedef __attribute__((ext_vector_type(16))) _Float16 v16h;
typedef __attribute__((ext_vector_type(8)))  _Float16 v8h;
typedef __attribute__((ext_vector_type(8)))  float    v8f;
typedef __attribute__((ext_vector_type(4)))  float    v4f;

#define N_NODES   131072
#define N_GRAPHS  4096
#define N_EDGES   524288
#define CH        96
#define NLAYERS   8
#define FEAT_IN   64
#define TOTAL_ROWS (N_NODES + N_GRAPHS)
#define GEMM_WAVES 8

__device__ __forceinline__ float silu_f(float x) {
    return x / (1.0f + __expf(-x));
}

// K index held by (lane, element j) of a v16h A/B fragment for
// V_WMMA_F32_16X16X32_F16 (CDNA5 ISA 7.12.2, 16-bit A 16x32 layout):
// lanes 0-15: j0..7 -> K=0..7,  j8..15 -> K=16..23
// lanes16-31: j0..7 -> K=8..15, j8..15 -> K=24..31
__device__ __forceinline__ int wmma_kmap(int lane, int j) {
    return ((j >= 8) ? 16 : 0) + ((lane >= 16) ? 8 : 0) + (j & 7);
}

// One-shot weight repack: W[K x NO] f32 row-major -> fragment-ordered f16.
// Wp layout: frag = kt*(NO/16)+nt; Wp[(frag*32 + lane)*16 + j] = W[k][n]
// so the GEMM's B-fragment load is one aligned 32-byte v16h load per lane.
__global__ void pack_w_kernel(const float* __restrict__ W,
                              _Float16* __restrict__ Wp,
                              int K, int NO)
{
    int idx = blockIdx.x * blockDim.x + threadIdx.x;
    if (idx >= K * NO) return;
    int j      = idx & 15;
    int lane   = (idx >> 4) & 31;
    int frag   = idx >> 9;
    int ntiles = NO >> 4;
    int kt = frag / ntiles, nt = frag - kt * ntiles;
    int k = kt * 32 + wmma_kmap(lane, j);
    int n = nt * 16 + (lane & 15);
    Wp[idx] = (_Float16)W[(size_t)k * NO + n];
}

// Fused WMMA GEMM:  Y[M x NO] = act(A[M x K]) @ W[K x NO] + bias
// AMODE: 0 = plain f32 A;  1 = f32 A + f32 A2 (h+agg);  2 = f16 A with BN+SiLU
// OMODE: 0 = store f32;    1 = store f16 + per-column sum/sumsq atomics
// K = KTILES*32 and NO = NTILES*16 are compile-time so all A/B/Y accesses
// are one base register + immediate offsets (no 64-bit address chains).
template<int AMODE, int OMODE, int KTILES, int NTILES>
__launch_bounds__(256)
__global__ void wmma_gemm_kernel(const void*  __restrict__ Aptr,
                                 const float* __restrict__ A2,
                                 const float* __restrict__ scaleK,
                                 const float* __restrict__ shiftK,
                                 const _Float16* __restrict__ Wp,
                                 const float* __restrict__ bias,
                                 void*  __restrict__ Yptr,
                                 float* __restrict__ stat_sum,
                                 float* __restrict__ stat_sq,
                                 int mtiles)
{
    constexpr int K  = KTILES * 32;
    constexpr int NO = NTILES * 16;
    const int lane = threadIdx.x & 31;
    const int tile = blockIdx.x * GEMM_WAVES + (threadIdx.x >> 5);
    if (tile >= mtiles) return;               // wave-uniform; EXEC stays full
    const int row0    = tile * 16;
    const int m       = row0 + (lane & 15);
    const int kb      = (lane >= 16) ? 8 : 0;
    const int halfsel = (lane >= 16) ? 8 : 0;

    union Frag { v16h v; _Float16 e[16]; };
    Frag afrag[KTILES];
    if (AMODE == 2) {
        const _Float16* Ab = (const _Float16*)Aptr + (size_t)m * K + kb;
        const float*    sc = scaleK + kb;
        const float*    sh = shiftK + kb;
        #pragma unroll
        for (int kt = 0; kt < KTILES; ++kt) {
            v8h r0 = *(const v8h*)(Ab + kt * 32);        // K = kt*32+kb ..+7
            v8h r1 = *(const v8h*)(Ab + kt * 32 + 16);   // K = kt*32+kb+16 ..
            #pragma unroll
            for (int j = 0; j < 8; ++j) {
                float t0 = (float)r0[j] * sc[kt * 32 + j]      + sh[kt * 32 + j];
                float t1 = (float)r1[j] * sc[kt * 32 + 16 + j] + sh[kt * 32 + 16 + j];
                afrag[kt].e[j]     = (_Float16)silu_f(t0);
                afrag[kt].e[8 + j] = (_Float16)silu_f(t1);
            }
        }
    } else {
        const float* Ab = (const float*)Aptr + (size_t)m * K + kb;
        #pragma unroll
        for (int kt = 0; kt < KTILES; ++kt) {
            v4f r0 = *(const v4f*)(Ab + kt * 32);
            v4f r1 = *(const v4f*)(Ab + kt * 32 + 4);
            v4f r2 = *(const v4f*)(Ab + kt * 32 + 16);
            v4f r3 = *(const v4f*)(Ab + kt * 32 + 20);
            if (AMODE == 1) {
                const float* A2b = A2 + (size_t)m * K + kb;
                r0 += *(const v4f*)(A2b + kt * 32);
                r1 += *(const v4f*)(A2b + kt * 32 + 4);
                r2 += *(const v4f*)(A2b + kt * 32 + 16);
                r3 += *(const v4f*)(A2b + kt * 32 + 20);
            }
            #pragma unroll
            for (int j = 0; j < 4; ++j) {
                afrag[kt].e[j]      = (_Float16)r0[j];
                afrag[kt].e[4 + j]  = (_Float16)r1[j];
                afrag[kt].e[8 + j]  = (_Float16)r2[j];
                afrag[kt].e[12 + j] = (_Float16)r3[j];
            }
        }
    }

    const int  ncol = lane & 15;
    const v16h* Wv  = (const v16h*)Wp;
    for (int nt = 0; nt < NTILES; ++nt) {
        v8f acc = {};
        #pragma unroll
        for (int kt = 0; kt < KTILES; ++kt) {
            Frag b;
            b.v = Wv[(size_t)(kt * NTILES + nt) * 32 + lane];  // 32B aligned
            acc = __builtin_amdgcn_wmma_f32_16x16x32_f16(
                false, afrag[kt].v, false, b.v, (short)0, acc, false, false);
        }
        // Epilogue: lane holds rows row0 + r + halfsel, column nt*16 + ncol.
        // NO is compile-time -> r*NO are immediate store offsets.
        const int col = nt * 16 + ncol;
        float bv = bias[col];
        float s = 0.f, sq = 0.f;
        if (OMODE == 0) {
            float* Yb = (float*)Yptr + (size_t)(row0 + halfsel) * NO + col;
            #pragma unroll
            for (int r = 0; r < 8; ++r) {
                Yb[r * NO] = acc[r] + bv;
            }
        } else {
            _Float16* Yb = (_Float16*)Yptr + (size_t)(row0 + halfsel) * NO + col;
            #pragma unroll
            for (int r = 0; r < 8; ++r) {
                float y = acc[r] + bv;
                Yb[r * NO] = (_Float16)y;
                s += y; sq += y * y;
            }
            // lanes l and l+16 hold the same column: combine, then 1 atomic.
            s  += __shfl_xor(s, 16, 32);
            sq += __shfl_xor(sq, 16, 32);
            if (lane < 16) {
                atomicAdd(&stat_sum[col], s);
                atomicAdd(&stat_sq[col],  sq);
            }
        }
    }
}

// h[N: N+G, :] = vn_emb (broadcast)
__global__ void fill_vn_kernel(float* __restrict__ h, const float* __restrict__ vn_emb)
{
    int idx = blockIdx.x * blockDim.x + threadIdx.x;
    if (idx >= N_GRAPHS * CH) return;
    h[(size_t)(N_NODES + idx / CH) * CH + (idx % CH)] = vn_emb[idx % CH];
}

// Segment-sum over the augmented edge list (E real edges + 2N virtual-node
// edges). One 96-lane block per edge; float atomics land in (192MB) L2.
__global__ void scatter_add_kernel(const float* __restrict__ h,
                                   float* __restrict__ agg,
                                   const int* __restrict__ edge_index,
                                   const int* __restrict__ batch)
{
    int e = blockIdx.x;
    int c = threadIdx.x;
    int s, t;
    if (e < N_EDGES)                { s = edge_index[e]; t = edge_index[N_EDGES + e]; }
    else if (e < N_EDGES + N_NODES) { int j = e - N_EDGES;           s = N_NODES + batch[j]; t = j; }
    else                            { int j = e - N_EDGES - N_NODES; s = j; t = N_NODES + batch[j]; }
    atomicAdd(&agg[(size_t)t * CH + c], h[(size_t)s * CH + c]);
}

// scale = g * rsqrt(var + eps); shift = beta - mean*scale  (population stats)
__global__ void bn_scale_kernel(const float* __restrict__ sum,
                                const float* __restrict__ sumsq,
                                const float* __restrict__ g,
                                const float* __restrict__ be,
                                float* __restrict__ scale,
                                float* __restrict__ shift,
                                int Cn, float invCount)
{
    int c = blockIdx.x * blockDim.x + threadIdx.x;
    if (c >= Cn) return;
    float m  = sum[c] * invCount;
    float v  = sumsq[c] * invCount - m * m;
    float sc = g[c] * rsqrtf(v + 1e-5f);
    scale[c] = sc;
    shift[c] = be[c] - m * sc;
}

// h = (first ? 0 : h) + silu(bn(z3))
__global__ void residual_apply_kernel(float* __restrict__ h,
                                      const void* __restrict__ z3p,
                                      const float* __restrict__ scale,
                                      const float* __restrict__ shift,
                                      int first)
{
    size_t idx = (size_t)blockIdx.x * blockDim.x + threadIdx.x;
    if (idx >= (size_t)TOTAL_ROWS * CH) return;
    int   c = (int)(idx % CH);
    float t = (float)((const _Float16*)z3p)[idx] * scale[c] + shift[c];
    float v = silu_f(t);
    h[idx] = (first ? 0.f : h[idx]) + v;
}

__global__ void pool_sum_kernel(const float* __restrict__ h,
                                const int* __restrict__ batch,
                                float* __restrict__ pooled,
                                float* __restrict__ counts)
{
    size_t idx = (size_t)blockIdx.x * blockDim.x + threadIdx.x;
    if (idx >= (size_t)N_NODES * CH) return;
    int i = (int)(idx / CH), c = (int)(idx % CH);
    int b = batch[i];
    atomicAdd(&pooled[(size_t)b * CH + c], h[idx]);
    if (c == 0) atomicAdd(&counts[b], 1.0f);
}

__global__ void pool_div_kernel(float* __restrict__ pooled,
                                const float* __restrict__ counts)
{
    int idx = blockIdx.x * blockDim.x + threadIdx.x;
    if (idx >= N_GRAPHS * CH) return;
    pooled[idx] /= fmaxf(counts[idx / CH], 1.0f);
}

// Tiny head GEMM (negligible FLOPs): one thread per output element.
template<int APPLY, int STATS>
__global__ void head_gemm_kernel(const float* __restrict__ X,
                                 const float* __restrict__ scale,
                                 const float* __restrict__ shift,
                                 const float* __restrict__ W,
                                 const float* __restrict__ bias,
                                 float* __restrict__ Y,
                                 float* __restrict__ ssum,
                                 float* __restrict__ ssq,
                                 int M, int K, int NO)
{
    int idx = blockIdx.x * blockDim.x + threadIdx.x;
    if (idx >= M * NO) return;
    int m = idx / NO, n = idx % NO;
    float acc = bias[n];
    for (int k = 0; k < K; ++k) {
        float x = X[(size_t)m * K + k];
        if (APPLY) { float t = x * scale[k] + shift[k]; x = silu_f(t); }
        acc += x * W[(size_t)k * NO + n];
    }
    Y[idx] = acc;
    if (STATS) { atomicAdd(&ssum[n], acc); atomicAdd(&ssq[n], acc * acc); }
}

extern "C" void kernel_launch(void* const* d_in, const int* in_sizes, int n_in,
                              void* d_out, int out_size, void* d_ws, size_t ws_size,
                              hipStream_t stream)
{
    (void)in_sizes; (void)n_in; (void)out_size; (void)ws_size;

    const float* x          = (const float*)d_in[0];
    const int*   edge_index = (const int*)  d_in[1];
    const int*   batch      = (const int*)  d_in[2];
    /* d_in[3] = num_graphs (compile-time constant N_GRAPHS) */
    const float* enc_w   = (const float*)d_in[4];
    const float* enc_b   = (const float*)d_in[5];
    const float* vn_emb  = (const float*)d_in[6];
    const float* vn_w1   = (const float*)d_in[7];
    const float* vn_b1   = (const float*)d_in[8];
    const float* vn_g1   = (const float*)d_in[9];
    const float* vn_be1  = (const float*)d_in[10];
    const float* vn_w2   = (const float*)d_in[11];
    const float* vn_b2   = (const float*)d_in[12];
    const float* conv_w1 = (const float*)d_in[13];
    const float* conv_b1 = (const float*)d_in[14];
    const float* conv_g1 = (const float*)d_in[15];
    const float* conv_be1= (const float*)d_in[16];
    const float* conv_w2 = (const float*)d_in[17];
    const float* conv_b2 = (const float*)d_in[18];
    const float* bn_g    = (const float*)d_in[19];
    const float* bn_b    = (const float*)d_in[20];
    const float* h_w1    = (const float*)d_in[21];
    const float* h_b1    = (const float*)d_in[22];
    const float* h_g1    = (const float*)d_in[23];
    const float* h_be1   = (const float*)d_in[24];
    const float* h_w2    = (const float*)d_in[25];
    const float* h_b2    = (const float*)d_in[26];
    const float* h_g2    = (const float*)d_in[27];
    const float* h_be2   = (const float*)d_in[28];
    const float* h_w3    = (const float*)d_in[29];
    const float* h_b3    = (const float*)d_in[30];
    float* out = (float*)d_out;

    // ---- workspace carve (256B aligned) ----
    size_t off = 0;
    char* ws_base = (char*)d_ws;
    auto take = [&](size_t bytes) -> void* {
        void* p = ws_base + off;
        off += (bytes + 255) & ~(size_t)255;
        return p;
    };
    float* h      = (float*)take((size_t)TOTAL_ROWS * CH * 4);      // f32 node state
    float* agg    = (float*)take((size_t)TOTAL_ROWS * CH * 4);      // f32 segment sums
    void*  z3     = (void*)agg;   // f16 z3 aliases agg (agg dead after GEMM1)
    void*  z2     = take((size_t)TOTAL_ROWS * 2 * CH * 2);          // f16 hidden (192ch)
    void*  vt     = take((size_t)N_GRAPHS * CH * 2);                // f16 vn hidden
    float* pooled = (float*)take((size_t)N_GRAPHS * CH * 4);
    float* counts = (float*)take((size_t)N_GRAPHS * 4);
    float* t1     = (float*)take((size_t)N_GRAPHS * 48 * 4);
    float* t2     = (float*)take((size_t)N_GRAPHS * 24 * 4);
    float* stats  = (float*)take(4096 * 4);
    // packed f16 weights (fragment order)
    _Float16* enc_wp  = (_Float16*)take((size_t)FEAT_IN * CH * 2);
    _Float16* vn_w1p  = (_Float16*)take((size_t)CH * CH * 2);
    _Float16* vn_w2p  = (_Float16*)take((size_t)CH * CH * 2);
    _Float16* conv_w1p= (_Float16*)take((size_t)NLAYERS * CH * 2 * CH * 2);
    _Float16* conv_w2p= (_Float16*)take((size_t)NLAYERS * 2 * CH * CH * 2);

    float* sum1   = stats + 0;    float* sq1    = stats + 192;
    float* scale1 = stats + 384;  float* shift1 = stats + 576;
    float* sum2   = stats + 768;  float* sq2    = stats + 864;
    float* scale2 = stats + 960;  float* shift2 = stats + 1056;
    float* vsum   = stats + 1152; float* vsq    = stats + 1248;
    float* vscale = stats + 1344; float* vshift = stats + 1440;
    float* hs1    = stats + 1536; float* hq1    = stats + 1584;
    float* hsc1   = stats + 1632; float* hsh1   = stats + 1680;
    float* hs2    = stats + 1728; float* hq2    = stats + 1752;
    float* hsc2   = stats + 1776; float* hsh2   = stats + 1800;

    const float invTotal = 1.0f / (float)TOTAL_ROWS;
    const float invG     = 1.0f / (float)N_GRAPHS;

    // ---- pack all weights to fragment-ordered f16 (tiny, L2-resident) ----
    pack_w_kernel<<<(FEAT_IN * CH + 255) / 256, 256, 0, stream>>>(enc_w, enc_wp, FEAT_IN, CH);
    pack_w_kernel<<<(CH * CH + 255) / 256, 256, 0, stream>>>(vn_w1, vn_w1p, CH, CH);
    pack_w_kernel<<<(CH * CH + 255) / 256, 256, 0, stream>>>(vn_w2, vn_w2p, CH, CH);
    for (int l = 0; l < NLAYERS; ++l) {
        pack_w_kernel<<<(CH * 2 * CH + 255) / 256, 256, 0, stream>>>(
            conv_w1 + (size_t)l * CH * 2 * CH, conv_w1p + (size_t)l * CH * 2 * CH, CH, 2 * CH);
        pack_w_kernel<<<(2 * CH * CH + 255) / 256, 256, 0, stream>>>(
            conv_w2 + (size_t)l * 2 * CH * CH, conv_w2p + (size_t)l * 2 * CH * CH, 2 * CH, CH);
    }

    // ---- encoder: h[:N] = x @ enc_w + enc_b  (K=64 -> 2 k-tiles) ----
    wmma_gemm_kernel<0, 0, 2, 6><<<(N_NODES / 16) / GEMM_WAVES, 256, 0, stream>>>(
        x, nullptr, nullptr, nullptr, enc_wp, enc_b, h, nullptr, nullptr, N_NODES / 16);
    fill_vn_kernel<<<(N_GRAPHS * CH + 255) / 256, 256, 0, stream>>>(h, vn_emb);

    const int n_edges_aug = N_EDGES + 2 * N_NODES;
    const int conv_grid   = (TOTAL_ROWS / 16) / GEMM_WAVES;   // 8448/8 = 1056
    const int vn_grid     = (N_GRAPHS / 16) / GEMM_WAVES;     // 256/8 = 32

    for (int l = 0; l < NLAYERS; ++l) {
        hipMemsetAsync(agg,  0, (size_t)TOTAL_ROWS * CH * 4, stream);
        hipMemsetAsync(sum1, 0, 384 * 4, stream);   // sum1+sq1
        hipMemsetAsync(sum2, 0, 192 * 4, stream);   // sum2+sq2

        scatter_add_kernel<<<n_edges_aug, CH, 0, stream>>>(h, agg, edge_index, batch);

        // z2 = (h + agg) @ W1 + b1   (96 -> 192), f16 out + BN stats
        wmma_gemm_kernel<1, 1, 3, 12><<<conv_grid, 256, 0, stream>>>(
            h, agg, nullptr, nullptr,
            conv_w1p + (size_t)l * CH * 2 * CH, conv_b1 + (size_t)l * 2 * CH,
            z2, sum1, sq1, TOTAL_ROWS / 16);
        bn_scale_kernel<<<1, 192, 0, stream>>>(
            sum1, sq1, conv_g1 + (size_t)l * 2 * CH, conv_be1 + (size_t)l * 2 * CH,
            scale1, shift1, 2 * CH, invTotal);

        // z3 = silu(bn(z2)) @ W2 + b2   (192 -> 96), f16 out + BN stats
        wmma_gemm_kernel<2, 1, 6, 6><<<conv_grid, 256, 0, stream>>>(
            z2, nullptr, scale1, shift1,
            conv_w2p + (size_t)l * 2 * CH * CH, conv_b2 + (size_t)l * CH,
            z3, sum2, sq2, TOTAL_ROWS / 16);
        bn_scale_kernel<<<1, 96, 0, stream>>>(
            sum2, sq2, bn_g + (size_t)l * CH, bn_b + (size_t)l * CH,
            scale2, shift2, CH, invTotal);

        // h = (l==0 ? 0 : h) + silu(bn(z3))
        residual_apply_kernel<<<(TOTAL_ROWS * CH + 255) / 256, 256, 0, stream>>>(
            h, z3, scale2, shift2, l == 0 ? 1 : 0);

        if (l < NLAYERS - 1) {
            hipMemsetAsync(vsum, 0, 192 * 4, stream);  // vsum+vsq
            // vt = h[n:] @ vn_w1 + vn_b1, f16 + BN stats over G rows
            wmma_gemm_kernel<0, 1, 3, 6><<<vn_grid, 256, 0, stream>>>(
                h + (size_t)N_NODES * CH, nullptr, nullptr, nullptr,
                vn_w1p, vn_b1, vt, vsum, vsq, N_GRAPHS / 16);
            bn_scale_kernel<<<1, 96, 0, stream>>>(
                vsum, vsq, vn_g1, vn_be1, vscale, vshift, CH, invG);
            // h[n:] = silu(bn(vt)) @ vn_w2 + vn_b2
            wmma_gemm_kernel<2, 0, 3, 6><<<vn_grid, 256, 0, stream>>>(
                vt, nullptr, vscale, vshift, vn_w2p, vn_b2,
                h + (size_t)N_NODES * CH, nullptr, nullptr, N_GRAPHS / 16);
        }
    }

    // ---- mean pool per graph ----
    hipMemsetAsync(pooled, 0, (size_t)N_GRAPHS * CH * 4, stream);
    hipMemsetAsync(counts, 0, (size_t)N_GRAPHS * 4, stream);
    hipMemsetAsync(hs1, 0, 96 * 4, stream);   // hs1+hq1
    hipMemsetAsync(hs2, 0, 48 * 4, stream);   // hs2+hq2
    pool_sum_kernel<<<((size_t)N_NODES * CH + 255) / 256, 256, 0, stream>>>(
        h, batch, pooled, counts);
    pool_div_kernel<<<(N_GRAPHS * CH + 255) / 256, 256, 0, stream>>>(pooled, counts);

    // ---- head MLP (tiny) ----
    head_gemm_kernel<0, 1><<<(N_GRAPHS * 48 + 255) / 256, 256, 0, stream>>>(
        pooled, nullptr, nullptr, h_w1, h_b1, t1, hs1, hq1, N_GRAPHS, CH, 48);
    bn_scale_kernel<<<1, 48, 0, stream>>>(hs1, hq1, h_g1, h_be1, hsc1, hsh1, 48, invG);

    head_gemm_kernel<1, 1><<<(N_GRAPHS * 24 + 255) / 256, 256, 0, stream>>>(
        t1, hsc1, hsh1, h_w2, h_b2, t2, hs2, hq2, N_GRAPHS, 48, 24);
    bn_scale_kernel<<<1, 24, 0, stream>>>(hs2, hq2, h_g2, h_be2, hsc2, hsh2, 24, invG);

    head_gemm_kernel<1, 0><<<(N_GRAPHS * 5 + 255) / 256, 256, 0, stream>>>(
        t2, hsc2, hsh2, h_w3, h_b3, out, nullptr, nullptr, N_GRAPHS, 24, 5);
}
